// SubjectLayersModel_31937376813637
// MI455X (gfx1250) — matrix operand order repaired and checked by reading, not compiled, not run on hardware
//
#include <hip/hip_runtime.h>

// ---------------------------------------------------------------------------
// Gathered batched GEMM:  out[b] = W[subjects[b]]^T @ x[b] + bias[subjects[b]]
//   x: (32, 512, 1024) f32, weights: (64, 512, 512) f32, bias: (64, 512) f32
//   out: (32, 512, 1024) f32.  M=D=512, N=T=1024, K=C=512 per batch.
// bf16 WMMA (v_wmma_f32_16x16x32_bf16), fp32 accumulation.
// HBM roofline ~8.2us (192MB @ 23.3TB/s); bf16 matrix ~7us -> balanced.
// Double-buffered LDS pipeline, one barrier per K-step.
// ---------------------------------------------------------------------------

typedef __attribute__((ext_vector_type(16))) __bf16 v16bf;
typedef __attribute__((ext_vector_type(2)))  __bf16 v2bf;
typedef __attribute__((ext_vector_type(8)))  float  v8f;

#define B_DIM   32
#define C_DIM   512
#define T_DIM   1024
#define D_DIM   512

#define WG_M    128            // D tile per workgroup
#define WG_N    128            // T tile per workgroup
#define K_STEP  32             // K per WMMA step
#define NSTEP   (C_DIM / K_STEP)
#define LDK     40             // padded k-stride (elements): 80B rows, 16B aligned
#define TILE_E  (WG_M * LDK)   // elements per tile buffer

// Pack two fp32 -> one packed bf16 pair as a single v_cvt_pk_bf16_f32.
__device__ __forceinline__ unsigned pk2(float lo, float hi) {
#if __has_builtin(__builtin_amdgcn_cvt_pk_bf16_f32)
  auto t = __builtin_amdgcn_cvt_pk_bf16_f32(lo, hi);
  static_assert(sizeof(t) == 4, "packed bf16 pair expected");
  union { decltype(t) v; unsigned u; } c; c.v = t;
  return c.u;
#else
  v2bf t = { (__bf16)lo, (__bf16)hi };   // fused to one packed convert
  union { v2bf v; unsigned u; } c; c.v = t;
  return c.u;
#endif
}

union FragAB { v16bf v; uint4 q[2]; };

__global__ __launch_bounds__(256)
void subject_gemm_wmma(const float* __restrict__ x,
                       const int*   __restrict__ subjects,
                       const float* __restrict__ weights,
                       const float* __restrict__ bias,
                       float*       __restrict__ out)
{
  // Tiles stored [column][k] (k contiguous, bf16), double buffered.
  __shared__ __align__(16) unsigned short As[2 * TILE_E]; // W tile: col = d
  __shared__ __align__(16) unsigned short Bs[2 * TILE_E]; // X tile: col = t
  __shared__ float biasS[WG_M];

  const int tid  = threadIdx.x;
  const int lane = tid & 31;
  const int wave = tid >> 5;      // 0..7
  const int wm   = wave & 1;      // 2 waves along M (D)
  const int wn   = wave >> 1;     // 4 waves along N (T)
  const int half = lane >> 4;
  const int m16  = lane & 15;

  const int tBlk = blockIdx.x * WG_N;
  const int dBlk = blockIdx.y * WG_M;
  const int b    = blockIdx.z;
  const int sb   = subjects[b];

  const float* Wb = weights + (size_t)sb * C_DIM * D_DIM + dBlk; // row c: +c*D
  const float* Xb = x       + (size_t)b  * C_DIM * T_DIM + tBlk; // row c: +c*T

  if (tid < WG_M) biasS[tid] = bias[(size_t)sb * D_DIM + dBlk + tid];

  v8f acc[4][2] = {};

  // Staging: thread covers 8 K-rows x 2 columns of each 32x128 fp32 tile.
  // Global: float2 per row, 64 consecutive threads span a full 512B row.
  // LDS: one b128 store per column (8 packed bf16), 2-way worst-case banks.
  const int kg = tid >> 6;          // 0..3 -> k-subchunk base = kg*8
  const int dg = (tid & 63) * 2;    // column pair

  const float* wptr = Wb + (size_t)(kg * 8) * D_DIM + dg;
  const float* xptr = Xb + (size_t)(kg * 8) * T_DIM + dg;

  float2 wa[8], xa[8];
  #pragma unroll
  for (int r = 0; r < 8; ++r) {
    wa[r] = *(const float2*)(wptr + (size_t)r * D_DIM);
    xa[r] = *(const float2*)(xptr + (size_t)r * T_DIM);
  }

  const int sBase = dg * LDK + kg * 8;          // staging store base (elements)

  // Fragment addressing (elements within one tile buffer):
  //  A (16x32 bf16): lane m / m+16 -> row m; K {0..7,16..23} / {8..15,24..31}
  //  B (32x16 bf16): lane n / n+16 -> col n; K {0..7,16..23} / {8..15,24..31}
  const int aRow0 = (wm * 64 + m16) * LDK;
  const int bRow0 = (wn * 32 + m16) * LDK;
  const int aLo = half ? 8 : 0, aHi = half ? 24 : 16;
  const int bLo = half * 8,     bHi = half * 8 + 16;

  for (int s = 0; s < NSTEP; ++s) {
    const int buf = (s & 1) * TILE_E;

    { // store staged registers -> LDS (fp32 -> bf16, transposed [col][k])
      unsigned short* A  = As + buf + sBase;
      unsigned short* Bp = Bs + buf + sBase;
      uint4 q;
      q.x = pk2(wa[0].x, wa[1].x); q.y = pk2(wa[2].x, wa[3].x);
      q.z = pk2(wa[4].x, wa[5].x); q.w = pk2(wa[6].x, wa[7].x);
      *(uint4*)(A) = q;
      q.x = pk2(wa[0].y, wa[1].y); q.y = pk2(wa[2].y, wa[3].y);
      q.z = pk2(wa[4].y, wa[5].y); q.w = pk2(wa[6].y, wa[7].y);
      *(uint4*)(A + LDK) = q;
      q.x = pk2(xa[0].x, xa[1].x); q.y = pk2(xa[2].x, xa[3].x);
      q.z = pk2(xa[4].x, xa[5].x); q.w = pk2(xa[6].x, xa[7].x);
      *(uint4*)(Bp) = q;
      q.x = pk2(xa[0].y, xa[1].y); q.y = pk2(xa[2].y, xa[3].y);
      q.z = pk2(xa[4].y, xa[5].y); q.w = pk2(xa[6].y, xa[7].y);
      *(uint4*)(Bp + LDK) = q;
    }
    __syncthreads();

    // Issue next K-tile global loads now; latency hidden behind WMMAs.
    if (s + 1 < NSTEP) {
      const float* wNext = wptr + (size_t)(s + 1) * K_STEP * D_DIM;
      const float* xNext = xptr + (size_t)(s + 1) * K_STEP * T_DIM;
      #pragma unroll
      for (int r = 0; r < 8; ++r) {
        wa[r] = *(const float2*)(wNext + (size_t)r * D_DIM);
        xa[r] = *(const float2*)(xNext + (size_t)r * T_DIM);
      }
    }

    // Fragment loads + 8 WMMAs.
    FragAB fa[4], fb[2];
    #pragma unroll
    for (int fm = 0; fm < 4; ++fm) {
      const unsigned short* p = As + buf + aRow0 + fm * 16 * LDK;
      fa[fm].q[0] = *(const uint4*)(p + aLo);
      fa[fm].q[1] = *(const uint4*)(p + aHi);
    }
    #pragma unroll
    for (int fn = 0; fn < 2; ++fn) {
      const unsigned short* p = Bs + buf + bRow0 + fn * 16 * LDK;
      fb[fn].q[0] = *(const uint4*)(p + bLo);
      fb[fn].q[1] = *(const uint4*)(p + bHi);
    }
    #pragma unroll
    for (int fm = 0; fm < 4; ++fm)
      #pragma unroll
      for (int fn = 0; fn < 2; ++fn)
        acc[fm][fn] = __builtin_amdgcn_wmma_f32_16x16x32_bf16(
            false, fa[fm].v, false, fb[fn].v,
            (short)0, acc[fm][fn], false, false);
  }

  // ---- epilogue: bias add + coalesced stores ----
  // C/D layout: VGPR i -> M = i (lanes 0-15) / i+8 (lanes 16-31), N = m16.
  #pragma unroll
  for (int fm = 0; fm < 4; ++fm) {
    const int dLoc = wm * 64 + fm * 16 + half * 8;
    #pragma unroll
    for (int fn = 0; fn < 2; ++fn) {
      const int tIdx = tBlk + wn * 32 + fn * 16 + m16;
      float* op = out + ((size_t)b * D_DIM + dBlk + dLoc) * T_DIM + tIdx;
      #pragma unroll
      for (int i = 0; i < 8; ++i)
        op[(size_t)i * T_DIM] = acc[fm][fn][i] + biasS[dLoc + i];
    }
  }
}

extern "C" void kernel_launch(void* const* d_in, const int* in_sizes, int n_in,
                              void* d_out, int out_size, void* d_ws, size_t ws_size,
                              hipStream_t stream) {
  const float* x        = (const float*)d_in[0];
  const int*   subjects = (const int*)  d_in[1];
  const float* weights  = (const float*)d_in[2];
  const float* bias     = (const float*)d_in[3];
  float*       out      = (float*)d_out;

  dim3 grid(T_DIM / WG_N, D_DIM / WG_M, B_DIM);  // (8, 4, 32)
  subject_gemm_wmma<<<grid, 256, 0, stream>>>(x, subjects, weights, bias, out);

  (void)in_sizes; (void)n_in; (void)out_size; (void)d_ws; (void)ws_size;
}
